// CGCN_22024592294341
// MI455X (gfx1250) — matrix operand order repaired and checked by reading, not compiled, not run on hardware
//
#include <hip/hip_runtime.h>
#include <hip/hip_bf16.h>

typedef __bf16 bf16_t;
typedef __attribute__((ext_vector_type(16))) __bf16 v16bf;
typedef __attribute__((ext_vector_type(8)))  float  v8f;

constexpr int Dc = 128;   // feature dim (D_IN == D == 128)
constexpr int Rr = 4;     // edge relations
constexpr int Tt = 4;     // node types
constexpr int Kf = 512;   // R*D fused K
constexpr int Cc = 8;     // classes

// 32-byte fragment <-> two 16-byte loads
union V16 {
  v16bf v;
  uint4 q[2];
};
union PK2 {
  bf16_t h[2];
  unsigned u;
};

// ---------------- precompute: bf16 weight conversion (TRANSPOSED [t][n][k]) -------
__global__ __launch_bounds__(256) void k_convert_whet_t(const float* __restrict__ W,
                                                        bf16_t* __restrict__ WbT) {
  int i = blockIdx.x * blockDim.x + threadIdx.x;   // (t, k, n), coalesced read
  if (i >= Tt * Dc * Dc) return;
  int n = i & (Dc - 1);
  int k = (i >> 7) & (Dc - 1);
  int t = i >> 14;
  WbT[((size_t)t * Dc + n) * Dc + k] = (bf16_t)W[i];
}

// M_T[l][n][r*128+k] = sum_j conv_W[l][r][k][j] * fuse_W[r*128+j][n]   (bf16 out)
__global__ __launch_bounds__(256) void k_compute_M_t(const float* __restrict__ convW,
                                                     const float* __restrict__ fuseW,
                                                     bf16_t* __restrict__ MbT, int L) {
  int i = blockIdx.x * blockDim.x + threadIdx.x;
  if (i >= L * Kf * Dc) return;
  int n  = i & (Dc - 1);
  int rk = (i >> 7) & (Kf - 1);
  int l  = i / (Kf * Dc);
  int r = rk >> 7, k = rk & (Dc - 1);
  const float* cw = convW + (((size_t)(l * Rr + r) * Dc + k) * Dc);
  float s = 0.f;
#pragma unroll 4
  for (int j = 0; j < Dc; ++j) s += cw[j] * fuseW[(size_t)(r * Dc + j) * Dc + n];
  MbT[((size_t)l * Dc + n) * Kf + rk] = (bf16_t)s;
}

// b_tot[l][n] = fuse_b[n] + sum_r sum_j conv_b[l][r][j] * fuse_W[r*128+j][n]
__global__ __launch_bounds__(256) void k_compute_btot(const float* __restrict__ convB,
                                                      const float* __restrict__ fuseW,
                                                      const float* __restrict__ fuseB,
                                                      float* __restrict__ btot, int L) {
  int i = blockIdx.x * blockDim.x + threadIdx.x;
  if (i >= L * Dc) return;
  int n = i & (Dc - 1);
  int l = i >> 7;
  float s = fuseB[n];
  for (int r = 0; r < Rr; ++r)
    for (int j = 0; j < Dc; ++j)
      s += convB[(size_t)(l * Rr + r) * Dc + j] * fuseW[(size_t)(r * Dc + j) * Dc + n];
  btot[i] = s;
}

// Stage a 64x32 f32->bf16 A tile into LDS: float2 loads, packed b32 LDS stores.
template <int LDA>
__device__ __forceinline__ void stage_A_tile(const float* __restrict__ A, int row0, int k0,
                                             int nrows, int tid, bf16_t* As) {
  unsigned* As32 = (unsigned*)As;
#pragma unroll
  for (int i = 0; i < 8; ++i) {
    int p = tid + i * 128;        // pair id 0..1023
    int r = p >> 4;               // 16 pairs (32 elems) per row
    int k = (p & 15) << 1;
    int gr = row0 + r;
    float2 v = make_float2(0.f, 0.f);
    if (gr < nrows) v = *reinterpret_cast<const float2*>(A + (size_t)gr * LDA + k0 + k);
    PK2 pk;
    pk.h[0] = (bf16_t)v.x;
    pk.h[1] = (bf16_t)v.y;
    As32[p] = pk.u;               // As[r][k..k+1]
  }
}

// A fragment (ISA 16-bit A layout): two contiguous 16B runs per lane.
__device__ __forceinline__ v16bf load_A_frag(const bf16_t* As, int arow, int half) {
  V16 a;
  const bf16_t* ap = As + arow * 32 + (half << 3);
  a.q[0] = *reinterpret_cast<const uint4*>(ap);        // K = half*8 + 0..7
  a.q[1] = *reinterpret_cast<const uint4*>(ap + 16);   // K = 16 + half*8 + 0..7
  return a.v;
}

// B fragment straight from transposed global weights [N][Ktot]: two 16-byte runs per lane.
template <int KTOT>
__device__ __forceinline__ v16bf load_B_frag(const bf16_t* __restrict__ WbT, int ncol, int k0,
                                             int half) {
  V16 b;
  const bf16_t* bp = WbT + (size_t)ncol * KTOT + k0 + (half << 3);
  b.q[0] = *reinterpret_cast<const uint4*>(bp);        // K = half*8 + 0..7
  b.q[1] = *reinterpret_cast<const uint4*>(bp + 16);   // K = 16 + half*8 + 0..7
  return b.v;
}

// ---------------- HeteroLinear: h = x @ W_het[type] + b_het[type] (WMMA) ----------
// 128 threads = 4 waves; block tile 64 rows x 128 cols; K = 128.
// A-fragment holds one row per lane (M = lane%16) -> per-type masking is a per-lane
// select of the A fragment vs zero; T masked WMMAs accumulate the exact result.
__global__ __launch_bounds__(128) void k_hetero_gemm(const float* __restrict__ X,
                                                     const bf16_t* __restrict__ WbT,  // [T][128N][128K]
                                                     const float* __restrict__ bhet,  // [T][128]
                                                     const int* __restrict__ ntype,
                                                     float* __restrict__ H, int nrows) {
  __shared__ __align__(16) bf16_t As[64 * 32];
  const int tid  = threadIdx.x;
  const int wave = tid >> 5;
  const int lane = tid & 31;
  const int half = lane >> 4;
  const int row0 = blockIdx.x * 64;
  const int wrow = row0 + wave * 16 + (lane & 15);
  const int trow = (wrow < nrows) ? ntype[wrow] : -1;

  v8f acc[8];
#pragma unroll
  for (int i = 0; i < 8; ++i)
#pragma unroll
    for (int e = 0; e < 8; ++e) acc[i][e] = 0.f;

  v16bf zf;
#pragma unroll
  for (int e = 0; e < 16; ++e) zf[e] = (bf16_t)0.f;

  for (int k0 = 0; k0 < Dc; k0 += 32) {
    stage_A_tile<Dc>(X, row0, k0, nrows, tid, As);
    __syncthreads();
    v16bf af = load_A_frag(As, wave * 16 + (lane & 15), half);
#pragma unroll
    for (int t = 0; t < Tt; ++t) {
      v16bf at = (trow == t) ? af : zf;   // per-lane type mask (lane == A row)
      const bf16_t* wt = WbT + (size_t)t * Dc * Dc;
#pragma unroll
      for (int nt = 0; nt < 8; ++nt) {
        v16bf bfr = load_B_frag<Dc>(wt, nt * 16 + (lane & 15), k0, half);
        acc[nt] = __builtin_amdgcn_wmma_f32_16x16x32_bf16(
            false, at, false, bfr, (short)0, acc[nt], false, false);
      }
    }
    __syncthreads();
  }
  // epilogue: + b_het[type[row]]   (C/D layout: row = v + half*8, col = lane%16)
#pragma unroll
  for (int nt = 0; nt < 8; ++nt)
#pragma unroll
    for (int v = 0; v < 8; ++v) {
      int gr = row0 + wave * 16 + v + (half << 3);
      int gc = nt * 16 + (lane & 15);
      if (gr < nrows) {
        int tt = ntype[gr];
        H[(size_t)gr * Dc + gc] = acc[nt][v] + bhet[(size_t)tt * Dc + gc];
      }
    }
}

// ---------------- scatter: agg[dst][r*128+d] += h[src][d], one wave per edge ------
__global__ __launch_bounds__(256) void k_scatter(const float* __restrict__ H,
                                                 const int* __restrict__ src,
                                                 const int* __restrict__ dst,
                                                 const int* __restrict__ etype,
                                                 float* __restrict__ agg, int E) {
  int g = blockIdx.x * blockDim.x + threadIdx.x;
  int e = g >> 5, lane = g & 31;
  if (e >= E) return;
  int s = src[e], d = dst[e], r = etype[e];
  float4 v = reinterpret_cast<const float4*>(H + (size_t)s * Dc)[lane];
  float* ap = agg + (size_t)d * Kf + r * Dc + lane * 4;
  atomicAdd(ap + 0, v.x);
  atomicAdd(ap + 1, v.y);
  atomicAdd(ap + 2, v.z);
  atomicAdd(ap + 3, v.w);
}

// ---------------- fused conv+fuse GEMM: h' = relu(agg[N,512] @ M + b_tot) ---------
__global__ __launch_bounds__(128) void k_fuse_gemm(const float* __restrict__ A,
                                                   const bf16_t* __restrict__ WbT,  // [128N][512K]
                                                   const float* __restrict__ bias,  // [128]
                                                   float* __restrict__ Out, int nrows) {
  __shared__ __align__(16) bf16_t As[64 * 32];
  const int tid  = threadIdx.x;
  const int wave = tid >> 5;
  const int lane = tid & 31;
  const int half = lane >> 4;
  const int row0 = blockIdx.x * 64;

  v8f acc[8];
#pragma unroll
  for (int i = 0; i < 8; ++i)
#pragma unroll
    for (int e = 0; e < 8; ++e) acc[i][e] = 0.f;

  for (int k0 = 0; k0 < Kf; k0 += 32) {
    stage_A_tile<Kf>(A, row0, k0, nrows, tid, As);
    __syncthreads();
    v16bf af = load_A_frag(As, wave * 16 + (lane & 15), half);
#pragma unroll
    for (int nt = 0; nt < 8; ++nt) {
      v16bf bfr = load_B_frag<Kf>(WbT, nt * 16 + (lane & 15), k0, half);
      acc[nt] = __builtin_amdgcn_wmma_f32_16x16x32_bf16(
          false, af, false, bfr, (short)0, acc[nt], false, false);
    }
    __syncthreads();
  }
#pragma unroll
  for (int nt = 0; nt < 8; ++nt)
#pragma unroll
    for (int v = 0; v < 8; ++v) {
      int gr = row0 + wave * 16 + v + (half << 3);
      int gc = nt * 16 + (lane & 15);
      if (gr < nrows)
        Out[(size_t)gr * Dc + gc] = fmaxf(acc[nt][v] + bias[gc], 0.f);
    }
}

// ---------------- global mean pool + classifier ----------------
__global__ __launch_bounds__(256) void k_pool_accum(const float* __restrict__ H,
                                                    const int* __restrict__ batch,
                                                    float* __restrict__ sum,
                                                    float* __restrict__ cnt, int nrows) {
  int g = blockIdx.x * blockDim.x + threadIdx.x;
  int n = g >> 5, lane = g & 31;
  if (n >= nrows) return;
  int b = batch[n];
  float4 v = reinterpret_cast<const float4*>(H + (size_t)n * Dc)[lane];
  float* sp = sum + (size_t)b * Dc + lane * 4;
  atomicAdd(sp + 0, v.x);
  atomicAdd(sp + 1, v.y);
  atomicAdd(sp + 2, v.z);
  atomicAdd(sp + 3, v.w);
  if (lane == 0) atomicAdd(cnt + b, 1.0f);
}

__global__ __launch_bounds__(256) void k_pool_final(const float* __restrict__ sum,
                                                    const float* __restrict__ cnt,
                                                    float* __restrict__ emb, int n) {
  int g = blockIdx.x * blockDim.x + threadIdx.x;
  if (g >= n) return;
  float c = fmaxf(cnt[g >> 7], 1.0f);
  emb[g] = sum[g] / c;
}

__global__ __launch_bounds__(256) void k_classify(const float* __restrict__ emb,
                                                  const float* __restrict__ clsW,  // [128][8]
                                                  const float* __restrict__ clsB,
                                                  float* __restrict__ logits, int G) {
  int g = blockIdx.x * blockDim.x + threadIdx.x;
  if (g >= G * Cc) return;
  int gr = g >> 3, c = g & 7;
  float s = clsB[c];
#pragma unroll 4
  for (int d = 0; d < Dc; ++d) s += emb[(size_t)gr * Dc + d] * clsW[d * Cc + c];
  logits[g] = s;
}

extern "C" void kernel_launch(void* const* d_in, const int* in_sizes, int n_in,
                              void* d_out, int out_size, void* d_ws, size_t ws_size,
                              hipStream_t stream) {
  (void)n_in; (void)ws_size;
  const float* x      = (const float*)d_in[0];
  const float* W_het  = (const float*)d_in[1];
  const float* b_het  = (const float*)d_in[2];
  const float* conv_W = (const float*)d_in[3];
  const float* conv_b = (const float*)d_in[4];
  const float* fuse_W = (const float*)d_in[5];
  const float* fuse_b = (const float*)d_in[6];
  const float* cls_W  = (const float*)d_in[7];
  const float* cls_b  = (const float*)d_in[8];
  const int* node_type  = (const int*)d_in[9];
  const int* edge_index = (const int*)d_in[10];
  const int* edge_type  = (const int*)d_in[11];
  const int* batch      = (const int*)d_in[12];

  const int N = in_sizes[9];
  const int E = in_sizes[11];
  const int G = out_size / (Dc + Cc);
  const int L = 2;
  const int* esrc = edge_index;
  const int* edst = edge_index + E;

  // workspace carve-out (256B aligned)
  char* ws = (char*)d_ws;
  size_t off = 0;
  auto carve = [&](size_t bytes) -> char* {
    char* p = ws + off;
    off += (bytes + 255) & ~(size_t)255;
    return p;
  };
  float*  h0    = (float*)carve((size_t)N * Dc * 4);
  float*  h1    = (float*)carve((size_t)N * Dc * 4);
  float*  agg   = (float*)carve((size_t)N * Kf * 4);
  bf16_t* WhetT = (bf16_t*)carve((size_t)Tt * Dc * Dc * 2);
  bf16_t* MbT   = (bf16_t*)carve((size_t)L * Dc * Kf * 2);
  float*  btot  = (float*)carve((size_t)L * Dc * 4);
  float*  psum  = (float*)carve((size_t)G * Dc * 4);
  float*  pcnt  = (float*)carve((size_t)G * 4);

  // fold linear weights (conv_W . fuse_W) and biases; bf16-convert, transposed [N][K]
  k_convert_whet_t<<<(Tt * Dc * Dc + 255) / 256, 256, 0, stream>>>(W_het, WhetT);
  k_compute_M_t<<<(L * Kf * Dc + 255) / 256, 256, 0, stream>>>(conv_W, fuse_W, MbT, L);
  k_compute_btot<<<(L * Dc + 255) / 256, 256, 0, stream>>>(conv_b, fuse_W, fuse_b, btot, L);

  const int gblocks = (N + 63) / 64;
  k_hetero_gemm<<<gblocks, 128, 0, stream>>>(x, WhetT, b_het, node_type, h0, N);

  float* hin = h0;
  float* hout = h1;
  for (int l = 0; l < L; ++l) {
    hipMemsetAsync(agg, 0, (size_t)N * Kf * 4, stream);
    int sblocks = (int)(((long)E * 32 + 255) / 256);
    k_scatter<<<sblocks, 256, 0, stream>>>(hin, esrc, edst, edge_type, agg, E);
    k_fuse_gemm<<<gblocks, 128, 0, stream>>>(agg, MbT + (size_t)l * Dc * Kf,
                                             btot + (size_t)l * Dc, hout, N);
    float* t = hin; hin = hout; hout = t;
  }

  hipMemsetAsync(psum, 0, (size_t)G * Dc * 4, stream);
  hipMemsetAsync(pcnt, 0, (size_t)G * 4, stream);
  k_pool_accum<<<(N * 32 + 255) / 256, 256, 0, stream>>>(hin, batch, psum, pcnt, N);

  float* emb = (float*)d_out;
  float* logits = emb + (size_t)G * Dc;
  k_pool_final<<<(G * Dc + 255) / 256, 256, 0, stream>>>(psum, pcnt, emb, G * Dc);
  k_classify<<<(G * Cc + 255) / 256, 256, 0, stream>>>(emb, cls_W, cls_b, logits, G);
}
